// MshmmLm_77541339562507
// MI455X (gfx1250) — matrix operand order repaired and checked by reading, compiled, not run on hardware
//
#include <hip/hip_runtime.h>
#include <hip/hip_bf16.h>
#include <math.h>

// ---------------------------------------------------------------------------
// MSHMM LM evidence on MI455X (gfx1250), wave32 + WMMA bf16.
// Pipeline:
//   1) 3x residual MLP (WMMA GEMMs 1024x512x512)
//   2) start logits + log_softmax (1024)
//   3) trans = R_state @ proj_w.T (WMMA 1024x1024x512) + row log_softmax
//   4) sparse term logits: only the 256K (word2state[v,k], v) pairs
//   5) Z[c] via scatter max / scatter sum-exp (dedup matches bool mask)
//   6) forward scan (255 steps, 8 states, per-batch block)
// GEMM: 32x32 tile per wave = 2x2 WMMA fragments, 4 v_wmma per K-step.
// Workspace: ~9.5 MB floats.
// ---------------------------------------------------------------------------

typedef __bf16 bf16_t;
typedef __attribute__((ext_vector_type(16))) __bf16 v16bf;
typedef __attribute__((ext_vector_type(8)))  float  v8f;

#define Bv   16
#define Tv   256
#define Vv   32000
#define Cv   1024
#define Hv   512
#define SPWv 8

// order-preserving float<->uint map for atomicMax on possibly-negative floats
__device__ __forceinline__ unsigned f2ord(float f) {
  unsigned u = __float_as_uint(f);
  return (u & 0x80000000u) ? ~u : (u | 0x80000000u);
}
__device__ __forceinline__ float ord2f(unsigned k) {
  unsigned u = (k & 0x80000000u) ? (k & 0x7FFFFFFFu) : ~k;
  return __uint_as_float(u);
}

// Load a 16x32 A-fragment (CDNA5 ISA 7.12.2 16-bit A layout):
//   lane m=L&15, half=L>>4; elems: K = half*8+i (i<8), 16+half*8+(i-8)
__device__ __forceinline__ v16bf load_a_frag(const float* __restrict__ row,
                                             int k0, int half) {
  v16bf a;
  const float* ap = row + k0 + half * 8;
#pragma unroll
  for (int i = 0; i < 8; ++i) {
    a[i]     = (bf16_t)ap[i];
    a[8 + i] = (bf16_t)ap[16 + i];
  }
  return a;
}

// Load a 32x16 B-fragment (lane n=L&15, half=L>>4; elems K = half*8+i (i<8), 16+half*8+(i-8))
__device__ __forceinline__ v16bf load_b_frag(const float* __restrict__ row,
                                             int k0, int half) {
  v16bf b;
  const float* bp = row + k0 + half * 8;
#pragma unroll
  for (int i = 0; i < 8; ++i) {
    b[i]     = (bf16_t)bp[i];
    b[8 + i] = (bf16_t)bp[16 + i];
  }
  return b;
}

// ---------------------------------------------------------------------------
// O[M,N] = act(A[M,K] @ W[N,K]^T + bias) (+ resid)
// One wave computes a 32x32 output tile: 2 A-frags x 2 B-frags -> 4 WMMAs/step.
// K multiple of 32, M/N multiples of 32.
// D layout: row = half*8 + r, col = L&15.
// ---------------------------------------------------------------------------
__global__ void gemm_wmma_bf16(const float* __restrict__ A,
                               const float* __restrict__ W,
                               const float* __restrict__ bias,
                               const float* __restrict__ resid,
                               float* __restrict__ O,
                               int M, int N, int K, int relu)
{
  const int n0   = blockIdx.x * 32;
  const int m0   = blockIdx.y * 32;
  const int lane = threadIdx.x & 31;
  const int idx  = lane & 15;
  const int half = lane >> 4;

  v8f acc00 = {}, acc01 = {}, acc10 = {}, acc11 = {};

  const float* arow0 = A + (size_t)(m0 + idx) * K;
  const float* arow1 = A + (size_t)(m0 + 16 + idx) * K;
  const float* wrow0 = W + (size_t)(n0 + idx) * K;
  const float* wrow1 = W + (size_t)(n0 + 16 + idx) * K;

  for (int k0 = 0; k0 < K; k0 += 32) {
    const v16bf a0 = load_a_frag(arow0, k0, half);
    const v16bf a1 = load_a_frag(arow1, k0, half);
    const v16bf b0 = load_b_frag(wrow0, k0, half);
    const v16bf b1 = load_b_frag(wrow1, k0, half);

    acc00 = __builtin_amdgcn_wmma_f32_16x16x32_bf16(false, a0, false, b0,
                                                    (short)0, acc00, false, false);
    acc01 = __builtin_amdgcn_wmma_f32_16x16x32_bf16(false, a0, false, b1,
                                                    (short)0, acc01, false, false);
    acc10 = __builtin_amdgcn_wmma_f32_16x16x32_bf16(false, a1, false, b0,
                                                    (short)0, acc10, false, false);
    acc11 = __builtin_amdgcn_wmma_f32_16x16x32_bf16(false, a1, false, b1,
                                                    (short)0, acc11, false, false);
  }

  const int col0 = n0 + idx;
  const int col1 = n0 + 16 + idx;
  const float bv0 = bias ? bias[col0] : 0.0f;
  const float bv1 = bias ? bias[col1] : 0.0f;

#pragma unroll
  for (int r = 0; r < 8; ++r) {
    const int row0 = m0 + half * 8 + r;
    const int row1 = m0 + 16 + half * 8 + r;

    float v00 = acc00[r] + bv0;
    float v01 = acc01[r] + bv1;
    float v10 = acc10[r] + bv0;
    float v11 = acc11[r] + bv1;
    if (relu) {
      v00 = fmaxf(v00, 0.0f); v01 = fmaxf(v01, 0.0f);
      v10 = fmaxf(v10, 0.0f); v11 = fmaxf(v11, 0.0f);
    }
    if (resid) {
      v00 += resid[(size_t)row0 * N + col0];
      v01 += resid[(size_t)row0 * N + col1];
      v10 += resid[(size_t)row1 * N + col0];
      v11 += resid[(size_t)row1 * N + col1];
    }
    O[(size_t)row0 * N + col0] = v00;
    O[(size_t)row0 * N + col1] = v01;
    O[(size_t)row1 * N + col0] = v10;
    O[(size_t)row1 * N + col1] = v11;
  }
}

// start logits: out[c] = dot(R[c], ow) + ob    (one wave per c)
__global__ void start_logits_kernel(const float* __restrict__ R,
                                    const float* __restrict__ ow,
                                    const float* __restrict__ ob,
                                    float* __restrict__ out)
{
  const int wid  = (int)((blockIdx.x * blockDim.x + threadIdx.x) >> 5);
  const int lane = threadIdx.x & 31;
  if (wid >= Cv) return;
  const float* r = R + (size_t)wid * Hv + lane * 16;
  const float* w = ow + lane * 16;
  float s = 0.f;
#pragma unroll
  for (int i = 0; i < 16; ++i) s += r[i] * w[i];
#pragma unroll
  for (int off = 16; off >= 1; off >>= 1) s += __shfl_xor(s, off, 32);
  if (lane == 0) out[wid] = s + ob[0];
}

// in-place row log_softmax; one block (256 thr) per row
__global__ void row_logsoftmax(float* __restrict__ data, int len)
{
  __shared__ float red[256];
  float* row = data + (size_t)blockIdx.x * len;
  const int tid = threadIdx.x;

  float m = -INFINITY;
  for (int i = tid; i < len; i += blockDim.x) m = fmaxf(m, row[i]);
  red[tid] = m; __syncthreads();
  for (int s = blockDim.x / 2; s > 0; s >>= 1) {
    if (tid < s) red[tid] = fmaxf(red[tid], red[tid + s]);
    __syncthreads();
  }
  m = red[0]; __syncthreads();

  float sum = 0.f;
  for (int i = tid; i < len; i += blockDim.x) sum += __expf(row[i] - m);
  red[tid] = sum; __syncthreads();
  for (int s = blockDim.x / 2; s > 0; s >>= 1) {
    if (tid < s) red[tid] += red[tid + s];
    __syncthreads();
  }
  const float lz = m + __logf(red[0]);
  __syncthreads();
  for (int i = tid; i < len; i += blockDim.x) row[i] -= lz;
}

// sparse term logits: pair[v,k] = dot(R[word2state[v,k]], term_ow[v]) + ob[v]
// one wave per word v
__global__ void pair_logits_kernel(const float* __restrict__ R,
                                   const float* __restrict__ ow,
                                   const float* __restrict__ ob,
                                   const int* __restrict__ w2s,
                                   float* __restrict__ pair)
{
  const int wid  = (int)((blockIdx.x * blockDim.x + threadIdx.x) >> 5);
  const int lane = threadIdx.x & 31;
  if (wid >= Vv) return;
  const float* w = ow + (size_t)wid * Hv + lane * 16;
  float wv[16];
#pragma unroll
  for (int i = 0; i < 16; ++i) wv[i] = w[i];
  const float bobs = ob[wid];
  for (int k = 0; k < SPWv; ++k) {
    const int c = w2s[wid * SPWv + k];
    const float* r = R + (size_t)c * Hv + lane * 16;
    float s = 0.f;
#pragma unroll
    for (int i = 0; i < 16; ++i) s += r[i] * wv[i];
#pragma unroll
    for (int off = 16; off >= 1; off >>= 1) s += __shfl_xor(s, off, 32);
    if (lane == 0) pair[wid * SPWv + k] = s + bobs;
  }
}

__global__ void init_z_kernel(unsigned* __restrict__ rmax, float* __restrict__ sexp)
{
  const int i = blockIdx.x * blockDim.x + threadIdx.x;
  if (i < Cv) { rmax[i] = 0u; sexp[i] = 0.f; }
}

__global__ void z_max_kernel(const float* __restrict__ pair,
                             const int* __restrict__ w2s,
                             unsigned* __restrict__ rmax)
{
  const int idx = blockIdx.x * blockDim.x + threadIdx.x;
  if (idx >= Vv * SPWv) return;
  const int v = idx / SPWv, k = idx % SPWv;
  const int c = w2s[idx];
  for (int kk = 0; kk < k; ++kk)           // dedup: mask is a set
    if (w2s[v * SPWv + kk] == c) return;
  atomicMax(&rmax[c], f2ord(pair[idx]));
}

__global__ void z_sum_kernel(const float* __restrict__ pair,
                             const int* __restrict__ w2s,
                             const unsigned* __restrict__ rmax,
                             float* __restrict__ sexp)
{
  const int idx = blockIdx.x * blockDim.x + threadIdx.x;
  if (idx >= Vv * SPWv) return;
  const int v = idx / SPWv, k = idx % SPWv;
  const int c = w2s[idx];
  for (int kk = 0; kk < k; ++kk)
    if (w2s[v * SPWv + kk] == c) return;
  atomicAdd(&sexp[c], __expf(pair[idx] - ord2f(rmax[c])));
}

__global__ void z_fin_kernel(const unsigned* __restrict__ rmax,
                             const float* __restrict__ sexp,
                             float* __restrict__ z)
{
  const int i = blockIdx.x * blockDim.x + threadIdx.x;
  if (i < Cv) z[i] = ord2f(rmax[i]) + __logf(sexp[i]);
}

// forward scan: one block (64 thr) per batch; thread = (n,j), 8-lane logsumexp
__global__ void scan_kernel(const int* __restrict__ text,
                            const int* __restrict__ w2s,
                            const float* __restrict__ trans,
                            const float* __restrict__ start_ls,
                            const float* __restrict__ pair,
                            const float* __restrict__ z,
                            float* __restrict__ blse)
{
  const int b   = blockIdx.x;
  const int tid = threadIdx.x;          // 64 threads
  const int n = tid >> 3, j = tid & 7;
  __shared__ float alpha[SPWv];
  __shared__ float anew[SPWv];
  if (tid < SPWv) alpha[tid] = 0.f;
  __syncthreads();

  const int* tb = text + b * Tv;
  for (int t = 0; t < Tv - 1; ++t) {
    const int vt = tb[t], vt1 = tb[t + 1];
    const int cj = w2s[vt * SPWv + j];    // previous state
    const int cn = w2s[vt1 * SPWv + n];   // next state
    float p = trans[(size_t)cj * Cv + cn];
    if (t == 0) p += start_ls[cj] + (pair[vt * SPWv + j] - z[cj]);
    p += pair[vt1 * SPWv + n] - z[cn];
    const float val = p + alpha[j];

    float m = val;
    m = fmaxf(m, __shfl_xor(m, 1, 8));
    m = fmaxf(m, __shfl_xor(m, 2, 8));
    m = fmaxf(m, __shfl_xor(m, 4, 8));
    float s = __expf(val - m);
    s += __shfl_xor(s, 1, 8);
    s += __shfl_xor(s, 2, 8);
    s += __shfl_xor(s, 4, 8);

    __syncthreads();
    if (j == 0) anew[n] = m + __logf(s);
    __syncthreads();
    if (tid < SPWv) alpha[tid] = anew[tid];
    __syncthreads();
  }

  if (tid == 0) {
    float m = alpha[0];
    for (int i = 1; i < SPWv; ++i) m = fmaxf(m, alpha[i]);
    float s = 0.f;
    for (int i = 0; i < SPWv; ++i) s += __expf(alpha[i] - m);
    blse[b] = m + __logf(s);
  }
}

__global__ void final_sum_kernel(const float* __restrict__ blse, float* __restrict__ out)
{
  if (threadIdx.x == 0 && blockIdx.x == 0) {
    float s = 0.f;
    for (int i = 0; i < Bv; ++i) s += blse[i];  // fixed order: deterministic
    out[0] = s;
  }
}

extern "C" void kernel_launch(void* const* d_in, const int* in_sizes, int n_in,
                              void* d_out, int out_size, void* d_ws, size_t ws_size,
                              hipStream_t stream)
{
  (void)in_sizes; (void)n_in; (void)out_size; (void)ws_size;

  const float* start_emb = (const float*)d_in[0];
  const float* start_l1w = (const float*)d_in[1];
  const float* start_l1b = (const float*)d_in[2];
  const float* start_l2w = (const float*)d_in[3];
  const float* start_l2b = (const float*)d_in[4];
  const float* start_ow  = (const float*)d_in[5];
  const float* start_ob  = (const float*)d_in[6];
  const float* state_emb = (const float*)d_in[7];
  const float* trans_l1w = (const float*)d_in[8];
  const float* trans_l1b = (const float*)d_in[9];
  const float* trans_l2w = (const float*)d_in[10];
  const float* trans_l2b = (const float*)d_in[11];
  const float* proj_w    = (const float*)d_in[12];
  const float* pret_emb  = (const float*)d_in[13];
  const float* term_l1w  = (const float*)d_in[14];
  const float* term_l1b  = (const float*)d_in[15];
  const float* term_l2w  = (const float*)d_in[16];
  const float* term_l2b  = (const float*)d_in[17];
  const float* term_ow   = (const float*)d_in[18];
  const float* term_ob   = (const float*)d_in[19];
  const int*   text      = (const int*)d_in[20];
  const int*   w2s       = (const int*)d_in[21];
  float* outp = (float*)d_out;

  // workspace layout (floats): ~9.43 MB total
  float* ws    = (float*)d_ws;
  float* HBUF  = ws;                    // 1024*512
  float* RBUF  = HBUF + Cv * Hv;        // 1024*512
  float* TRANS = RBUF + Cv * Hv;        // 1024*1024
  float* SBUF  = TRANS + Cv * Cv;       // 1024
  float* PAIR  = SBUF + Cv;             // 32000*8
  unsigned* RMAX = (unsigned*)(PAIR + Vv * SPWv);  // 1024
  float* SEXP  = (float*)(RMAX + Cv);   // 1024
  float* ZB    = SEXP + Cv;             // 1024
  float* BLSE  = ZB + Cv;               // 16

  const dim3 blk32(32);
  const dim3 gHH(Hv / 32, Cv / 32);     // 1024x512 output, 32x32 tiles
  const dim3 gCC(Cv / 32, Cv / 32);     // 1024x1024 output, 32x32 tiles

  // residual MLP: RBUF = X + relu(relu(X@W1.T+b1)@W2.T + b2)
  auto mlp = [&](const float* X, const float* w1, const float* b1,
                 const float* w2, const float* b2) {
    gemm_wmma_bf16<<<gHH, blk32, 0, stream>>>(X, w1, b1, nullptr, HBUF,
                                              Cv, Hv, Hv, /*relu=*/1);
    gemm_wmma_bf16<<<gHH, blk32, 0, stream>>>(HBUF, w2, b2, X, RBUF,
                                              Cv, Hv, Hv, /*relu=*/1);
  };

  // 1) start distribution
  mlp(start_emb, start_l1w, start_l1b, start_l2w, start_l2b);
  start_logits_kernel<<<dim3(Cv * 32 / 256), dim3(256), 0, stream>>>(
      RBUF, start_ow, start_ob, SBUF);
  row_logsoftmax<<<dim3(1), dim3(256), 0, stream>>>(SBUF, Cv);

  // 2) transition matrix
  mlp(state_emb, trans_l1w, trans_l1b, trans_l2w, trans_l2b);
  gemm_wmma_bf16<<<gCC, blk32, 0, stream>>>(RBUF, proj_w, nullptr, nullptr,
                                            TRANS, Cv, Cv, Hv, /*relu=*/0);
  row_logsoftmax<<<dim3(Cv), dim3(256), 0, stream>>>(TRANS, Cv);

  // 3) sparse emission logits + per-state normalizers
  mlp(pret_emb, term_l1w, term_l1b, term_l2w, term_l2b);
  pair_logits_kernel<<<dim3(Vv * 32 / 256), dim3(256), 0, stream>>>(
      RBUF, term_ow, term_ob, w2s, PAIR);
  init_z_kernel<<<dim3((Cv + 255) / 256), dim3(256), 0, stream>>>(RMAX, SEXP);
  z_max_kernel<<<dim3((Vv * SPWv + 255) / 256), dim3(256), 0, stream>>>(PAIR, w2s, RMAX);
  z_sum_kernel<<<dim3((Vv * SPWv + 255) / 256), dim3(256), 0, stream>>>(PAIR, w2s, RMAX, SEXP);
  z_fin_kernel<<<dim3((Cv + 255) / 256), dim3(256), 0, stream>>>(RMAX, SEXP, ZB);

  // 4) forward scan + evidence
  scan_kernel<<<dim3(Bv), dim3(64), 0, stream>>>(text, w2s, TRANS, SBUF, PAIR, ZB, BLSE);
  final_sum_kernel<<<dim3(1), dim3(32), 0, stream>>>(BLSE, outp);
}
